// MultiClassContrastiveLoss_78185584657074
// MI455X (gfx1250) — matrix operand order repaired and checked by reading, not compiled, not run on hardware
//
#include <hip/hip_runtime.h>
#include <hip/hip_bf16.h>

typedef __attribute__((ext_vector_type(16))) _Float16 v16h;
typedef __attribute__((ext_vector_type(8)))  _Float16 v8h;
typedef __attribute__((ext_vector_type(8)))  float    v8f;

#define DDIM 256
#define NCLS 1000
#define CPAD 1008            // 63 tiles of 16; padded classes zeroed + masked
#define NTILES (CPAD / 16)   // 63
#define NORM_EPS 1e-12f
#define COS_EPS  1e-8f

// ---------------------------------------------------------------- init ------
__global__ __launch_bounds__(256) void mccl_init_kernel(int* __restrict__ counts) {
  int i = blockIdx.x * blockDim.x + threadIdx.x;
  if (i < NCLS) counts[i] = 0;
}

// ------------------------------------------------- normalize + histogram ----
// One wave per row; lane handles 8 strided elements. Writes f16 normalized copy.
__global__ __launch_bounds__(256) void mccl_normalize_kernel(
    const float* __restrict__ emb, const int* __restrict__ labels,
    _Float16* __restrict__ eh, int* __restrict__ counts, int nrows) {
  int wave = threadIdx.x >> 5, lane = threadIdx.x & 31;
  int row = blockIdx.x * 8 + wave;
  if (row >= nrows) return;
  const float* x = emb + (size_t)row * DDIM;
  float v[8];
  float ss = 0.0f;
#pragma unroll
  for (int i = 0; i < 8; ++i) { v[i] = x[lane + i * 32]; ss += v[i] * v[i]; }
#pragma unroll
  for (int o = 16; o > 0; o >>= 1) ss += __shfl_xor(ss, o, 32);
  float inv = 1.0f / fmaxf(sqrtf(ss), NORM_EPS);
  _Float16* eo = eh + (size_t)row * DDIM;
#pragma unroll
  for (int i = 0; i < 8; ++i) eo[lane + i * 32] = (_Float16)(v[i] * inv);
  if (lane == 0) atomicAdd(&counts[labels[row]], 1);
}

// ------------------------------------------------------------ prefix scan ---
// One block, 1024 threads: exclusive offsets over 1000 counts; cursors copy.
__global__ __launch_bounds__(1024) void mccl_scan_kernel(
    const int* __restrict__ counts, int* __restrict__ offsets,
    int* __restrict__ cursors) {
  __shared__ int tmp[1024];
  int t = threadIdx.x;
  int v = (t < NCLS) ? counts[t] : 0;
  tmp[t] = v;
  __syncthreads();
#pragma unroll
  for (int o = 1; o < 1024; o <<= 1) {
    int add = (t >= o) ? tmp[t - o] : 0;
    __syncthreads();
    tmp[t] += add;
    __syncthreads();
  }
  if (t < NCLS) {
    int excl = tmp[t] - v;
    offsets[t] = excl;
    cursors[t] = excl;
    if (t == NCLS - 1) offsets[NCLS] = tmp[t];
  }
}

// ---------------------------------------------------------------- scatter ---
__global__ __launch_bounds__(256) void mccl_scatter_kernel(
    const int* __restrict__ labels, int* __restrict__ cursors,
    int* __restrict__ order, int nrows) {
  int i = blockIdx.x * blockDim.x + threadIdx.x;
  if (i >= nrows) return;
  int pos = atomicAdd(&cursors[labels[i]], 1);
  order[pos] = i;
}

// ---------------------------------------- per-class gather-sum -> centers ---
// Block per class; thread d owns dimension d. Coalesced 512B reads per row.
__global__ __launch_bounds__(256) void mccl_centers_kernel(
    const _Float16* __restrict__ eh, const int* __restrict__ order,
    const int* __restrict__ offsets, _Float16* __restrict__ chh,
    float* __restrict__ inv_denom) {
  int c = blockIdx.x, d = threadIdx.x;
  if (c >= NCLS) {  // zero padding rows
    chh[(size_t)c * DDIM + d] = (_Float16)0.0f;
    if (d == 0) inv_denom[c] = 0.0f;
    return;
  }
  int beg = offsets[c], end = offsets[c + 1];
  float s0 = 0.f, s1 = 0.f, s2 = 0.f, s3 = 0.f;
  int j = beg;
  for (; j + 4 <= end; j += 4) {
    s0 += (float)eh[(size_t)order[j + 0] * DDIM + d];
    s1 += (float)eh[(size_t)order[j + 1] * DDIM + d];
    s2 += (float)eh[(size_t)order[j + 2] * DDIM + d];
    s3 += (float)eh[(size_t)order[j + 3] * DDIM + d];
  }
  for (; j < end; ++j) s0 += (float)eh[(size_t)order[j] * DDIM + d];
  float s = (s0 + s1) + (s2 + s3);
  float cnt = (float)(end - beg);
  float ctr = s / fmaxf(cnt, 1.0f);
  __shared__ float red[256];
  red[d] = ctr * ctr;
  __syncthreads();
#pragma unroll
  for (int o = 128; o > 0; o >>= 1) {
    if (d < o) red[d] += red[d + o];
    __syncthreads();
  }
  float invd = 1.0f / fmaxf(sqrtf(red[0]), COS_EPS);
  chh[(size_t)c * DDIM + d] = (_Float16)ctr;
  if (d == 0) inv_denom[c] = invd;
}

// ------------------------------------------------------- fused GEMM+loss ----
// 8 waves/block; wave owns 16 rows. A strip register-resident; B tiles
// ping-pong double-buffered so loads for tile nt+1 overlap WMMAs of tile nt.
#define LOAD_B(NT_, B_, INVD_)                                                 \
  {                                                                            \
    const _Float16* brow_ = chh + (size_t)((NT_)*16 + l16) * DDIM + half * 8;  \
    _Pragma("unroll") for (int kc = 0; kc < 8; ++kc) {                         \
      v8h b0_ = *(const v8h*)(brow_ + kc * 32);                                \
      v8h b1_ = *(const v8h*)(brow_ + kc * 32 + 16);                           \
      B_[kc] = __builtin_shufflevector(b0_, b1_, 0, 1, 2, 3, 4, 5, 6, 7, 8, 9, \
                                       10, 11, 12, 13, 14, 15);                \
    }                                                                          \
    INVD_ = inv_denom[(NT_)*16 + l16];                                         \
  }

#define COMPUTE(NT_, B_, INVD_)                                                \
  {                                                                            \
    v8f acc = {};                                                              \
    _Pragma("unroll") for (int kc = 0; kc < 8; ++kc) {                         \
      acc = __builtin_amdgcn_wmma_f32_16x16x32_f16(false, a[kc], false,        \
                                                   B_[kc], (short)0, acc,      \
                                                   false, false);              \
    }                                                                          \
    int col = (NT_)*16 + l16;                                                  \
    float mask = (col < NCLS) ? 1.0f : 0.0f;                                   \
    _Pragma("unroll") for (int r = 0; r < 8; ++r) {                            \
      float sim = acc[r] * INVD_;                                              \
      float u = 1.0f - sim;                                                    \
      float relu = fmaxf(u, 0.0f);                                             \
      bool isl = (col == labs[r]);                                             \
      p += mask * (isl ? u * u : 0.0f);                                        \
      g += mask * (isl ? 0.0f : relu * relu);                                  \
    }                                                                          \
  }

__global__ __launch_bounds__(256) void mccl_gemm_loss_kernel(
    const _Float16* __restrict__ eh, const _Float16* __restrict__ chh,
    const float* __restrict__ inv_denom, const int* __restrict__ labels,
    float* __restrict__ partial, int nrows) {
  int wave = threadIdx.x >> 5, lane = threadIdx.x & 31;
  int l16 = lane & 15, half = lane >> 4;
  int row_base = blockIdx.x * 128 + wave * 16;

  float p = 0.0f, g = 0.0f;
  if (row_base + 16 <= nrows) {
    // A fragments, ISA 16-bit A 16x32 layout:
    //  low half lanes:  v0-3 K=k0+0..7,  v4-7 K=k0+16..23
    //  high half lanes: v0-3 K=k0+8..15, v4-7 K=k0+24..31
    const _Float16* arow = eh + (size_t)(row_base + l16) * DDIM;
    v16h a[8];
#pragma unroll
    for (int kc = 0; kc < 8; ++kc) {
      v8h lo = *(const v8h*)(arow + kc * 32 + half * 8);
      v8h hi = *(const v8h*)(arow + kc * 32 + 16 + half * 8);
      a[kc] = __builtin_shufflevector(lo, hi, 0, 1, 2, 3, 4, 5, 6, 7, 8, 9, 10,
                                      11, 12, 13, 14, 15);
    }
    int labs[8];
#pragma unroll
    for (int r = 0; r < 8; ++r) labs[r] = labels[row_base + half * 8 + r];

    v16h b0buf[8], b1buf[8];
    float invd0, invd1;
    LOAD_B(0, b0buf, invd0);
    for (int nt = 0; nt < NTILES; nt += 2) {
      if (nt + 1 < NTILES) LOAD_B(nt + 1, b1buf, invd1);
      COMPUTE(nt, b0buf, invd0);
      if (nt + 2 < NTILES) LOAD_B(nt + 2, b0buf, invd0);
      if (nt + 1 < NTILES) COMPUTE(nt + 1, b1buf, invd1);
    }
  }

  float local = p + g * (1.0f / (float)(NCLS - 1));
  __shared__ float red[256];
  red[threadIdx.x] = local;
  __syncthreads();
#pragma unroll
  for (int s = 128; s > 0; s >>= 1) {
    if (threadIdx.x < s) red[threadIdx.x] += red[threadIdx.x + s];
    __syncthreads();
  }
  if (threadIdx.x == 0) partial[blockIdx.x] = red[0];
}

// --------------------------------------------------------------- finalize ---
__global__ __launch_bounds__(256) void mccl_finalize_kernel(
    const float* __restrict__ partial, float* __restrict__ out, int nblk,
    int nrows) {
  __shared__ float red[256];
  float s = 0.0f;
  for (int i = threadIdx.x; i < nblk; i += 256) s += partial[i];
  red[threadIdx.x] = s;
  __syncthreads();
#pragma unroll
  for (int o = 128; o > 0; o >>= 1) {
    if (threadIdx.x < o) red[threadIdx.x] += red[threadIdx.x + o];
    __syncthreads();
  }
  if (threadIdx.x == 0) out[0] = red[0] / (float)nrows;
}

// ----------------------------------------------------------------- launch ---
extern "C" void kernel_launch(void* const* d_in, const int* in_sizes, int n_in,
                              void* d_out, int out_size, void* d_ws, size_t ws_size,
                              hipStream_t stream) {
  const float* emb   = (const float*)d_in[0];
  const int*  labels = (const int*)d_in[1];
  float*      out    = (float*)d_out;
  int nrows = in_sizes[1];          // 65536
  int nblk  = nrows / 128;          // 512 GEMM blocks

  char* ws = (char*)d_ws;
  size_t off = 0;
  int* counts     = (int*)(ws + off);   off += (size_t)NCLS * sizeof(int);
  int* offsets    = (int*)(ws + off);   off += (size_t)(NCLS + 1) * sizeof(int);
  int* cursors    = (int*)(ws + off);   off += (size_t)NCLS * sizeof(int);
  int* order      = (int*)(ws + off);   off += (size_t)nrows * sizeof(int);
  float* inv_denom= (float*)(ws + off); off += (size_t)CPAD * sizeof(float);
  float* partial  = (float*)(ws + off); off += (size_t)nblk * sizeof(float);
  off = (off + 255) & ~(size_t)255;
  _Float16* eh  = (_Float16*)(ws + off); off += (size_t)nrows * DDIM * sizeof(_Float16);
  off = (off + 255) & ~(size_t)255;
  _Float16* chh = (_Float16*)(ws + off); off += (size_t)CPAD * DDIM * sizeof(_Float16);
  (void)ws_size; (void)n_in; (void)out_size;

  mccl_init_kernel<<<(NCLS + 255) / 256, 256, 0, stream>>>(counts);

  mccl_normalize_kernel<<<(nrows + 7) / 8, 256, 0, stream>>>(
      emb, labels, eh, counts, nrows);

  mccl_scan_kernel<<<1, 1024, 0, stream>>>(counts, offsets, cursors);

  mccl_scatter_kernel<<<(nrows + 255) / 256, 256, 0, stream>>>(
      labels, cursors, order, nrows);

  mccl_centers_kernel<<<CPAD, 256, 0, stream>>>(eh, order, offsets, chh,
                                                inv_denom);

  mccl_gemm_loss_kernel<<<nblk, 256, 0, stream>>>(eh, chh, inv_denom, labels,
                                                  partial, nrows);

  mccl_finalize_kernel<<<1, 256, 0, stream>>>(partial, out, nblk, nrows);
}
